// MultiQueryAttention_57672820850754
// MI455X (gfx1250) — compile-verified, matrix operand order checked by reading, not run on hardware
//
#include <hip/hip_runtime.h>
#include <hip/hip_bf16.h>

// ---------------------------------------------------------------------------
// MQA forward for gfx1250 (MI455X): bf16 WMMA everywhere, fp32 accumulate.
//   B=4, S=2048, D_MODEL=2048, H=16, HEAD_DIM=128  (shapes hardcoded)
// Pipeline:
//   1) convert hs / weights fp32 -> bf16 (qkv weights fused into 2304x2048)
//   2) QKV GEMM   (8192 x 2304 x 2048)  -> bf16 QKV buffer
//      - 128x256 block tile, 64x64 wave tile (16 WMMA / 8 frag-loads / k-step)
//      - double-buffered LDS fed by GLOBAL_LOAD_ASYNC_TO_LDS_B128 (ASYNCcnt)
//   3) flash MQA  (causal, online softmax, WMMA for QK^T and PV)
//   4) O GEMM     (8192 x 2048 x 2048)  -> fp32 d_out (+ bias)
// Workspace use: ~123 MB.
// ---------------------------------------------------------------------------

typedef __attribute__((ext_vector_type(16))) __bf16 v16bf;
typedef __attribute__((ext_vector_type(8)))  float  v8f;
typedef int v4i_b128 __attribute__((vector_size(16)));   // matches builtin

union FragU {
    v16bf v;
    uint4 q[2];
};

__device__ __forceinline__ v8f v8f_zero() {
    v8f z;
#pragma unroll
    for (int i = 0; i < 8; ++i) z[i] = 0.f;
    return z;
}

__device__ __forceinline__ unsigned short f32_to_bf16(float f) {
    unsigned int u = __float_as_uint(f);
    u = (u + 0x7FFFu + ((u >> 16) & 1u)) >> 16;   // round-to-nearest-even
    return (unsigned short)u;
}

// ---------------------------------------------------------------------------
// Async global -> LDS copy (ASYNCcnt path). Builtin prototype (from compiler
// diagnostics): (v4i __device__* src, v4i __shared__* dst, Ii off, Ii cpol).
// ---------------------------------------------------------------------------
#if defined(__has_builtin)
#if __has_builtin(__builtin_amdgcn_global_load_async_to_lds_b128)
#define HAVE_ASYNC_LDS 1
#endif
#endif
#ifndef HAVE_ASYNC_LDS
#define HAVE_ASYNC_LDS 0
#endif

typedef __attribute__((address_space(1))) v4i_b128* as1_v4i_ptr;
typedef __attribute__((address_space(3))) v4i_b128* as3_v4i_ptr;

__device__ __forceinline__ void async_copy_b128(const unsigned short* g,
                                                unsigned short* l) {
#if HAVE_ASYNC_LDS
    __builtin_amdgcn_global_load_async_to_lds_b128(
        (as1_v4i_ptr)g, (as3_v4i_ptr)l, /*offset=*/0, /*cpol=*/0);
#else
    *(uint4*)l = *(const uint4*)g;
#endif
}

__device__ __forceinline__ void async_wait_all() {
#if HAVE_ASYNC_LDS
    asm volatile("s_wait_asynccnt 0" ::: "memory");
#endif
}

// Load a 16x32 bf16 WMMA A/B fragment from a row-major tile.
// tile points at (row 0, col 0); ld = row stride in elements.
// Per ISA: lanes 0-15 row=lane, K = {c0..c0+7, c0+16..c0+23}, c0 = 8*(lane>>4).
__device__ __forceinline__ v16bf load_frag(const unsigned short* tile, int ld) {
    const int lane = threadIdx.x & 31;
    const int r    = lane & 15;
    const int c0   = (lane >> 4) << 3;
    const unsigned short* p = tile + r * ld + c0;
    FragU u;
    u.q[0] = *(const uint4*)(p);
    u.q[1] = *(const uint4*)(p + 16);
    return u.v;
}

__device__ __forceinline__ v8f wmma_bf16(v16bf a, v16bf b, v8f c) {
    return __builtin_amdgcn_wmma_f32_16x16x32_bf16(
        /*neg_a=*/false, a, /*neg_b=*/false, b,
        /*c_mod=*/(short)0, c, /*reuse_a=*/false, /*reuse_b=*/false);
}

// ---------------------------------------------------------------------------
// fp32 -> bf16 elementwise conversion
// ---------------------------------------------------------------------------
__global__ __launch_bounds__(256) void convert_f32_bf16_kernel(
    const float* __restrict__ src, unsigned short* __restrict__ dst, long n) {
    long i      = (long)blockIdx.x * blockDim.x + threadIdx.x;
    long stride = (long)gridDim.x * blockDim.x;
    for (; i < n; i += stride) dst[i] = f32_to_bf16(src[i]);
}

// ---------------------------------------------------------------------------
// GEMM: C[M,N] = A[M,K](bf16) * W[N,K]^T(bf16) + bias[N]
// Block tile 128(M) x 256(N), K-step 32, double-buffered async LDS staging.
// 256 threads = 8 waves in 2(M)x4(N); each wave owns 64x64 = 4x4 accumulators.
// ---------------------------------------------------------------------------
template <bool OUT_BF16>
__global__ __launch_bounds__(256) void gemm_bf16_kernel(
    const unsigned short* __restrict__ A,   // M x K
    const unsigned short* __restrict__ W,   // N x K
    const float* __restrict__ bias,         // N
    void* __restrict__ Cout,                // M x ldc
    int M, int N, int K, int ldc) {
    constexpr int TK  = 32;
    constexpr int LDT = 40;                 // padded LDS row stride (80B)
    constexpr int BM  = 128;
    constexpr int BN  = 256;
    __shared__ unsigned short As[2][BM * LDT];   // 2 x 10 KB
    __shared__ unsigned short Bs[2][BN * LDT];   // 2 x 20 KB

    const int tid  = threadIdx.x;
    const int wave = tid >> 5;
    const int wm   = wave >> 2;             // 0..1  (64-row strip)
    const int wn   = wave & 3;              // 0..3  (64-col strip)
    const int m0   = blockIdx.y * BM;
    const int n0   = blockIdx.x * BN;

    v8f acc[4][4];
#pragma unroll
    for (int i = 0; i < 4; ++i)
#pragma unroll
        for (int j = 0; j < 4; ++j) acc[i][j] = v8f_zero();

    // Issue one 32-wide K slab of A (128x32) + W (256x32) into LDS buffer buf.
    auto issue_tile = [&](int k0, int buf) {
#pragma unroll
        for (int i = 0; i < 2; ++i) {       // A: 512 uint4 / 256 threads
            int idx = tid + i * 256;
            int row = idx >> 2;
            int c8  = (idx & 3) << 3;
            async_copy_b128(&A[(long)(m0 + row) * K + k0 + c8],
                            &As[buf][row * LDT + c8]);
        }
#pragma unroll
        for (int i = 0; i < 4; ++i) {       // W: 1024 uint4 / 256 threads
            int idx = tid + i * 256;
            int row = idx >> 2;
            int c8  = (idx & 3) << 3;
            async_copy_b128(&W[(long)(n0 + row) * K + k0 + c8],
                            &Bs[buf][row * LDT + c8]);
        }
    };

    issue_tile(0, 0);
    async_wait_all();
    __syncthreads();

    const int nk = K / TK;
    for (int t = 0; t < nk; ++t) {
        const int buf = t & 1;
        if (t + 1 < nk) issue_tile((t + 1) * TK, buf ^ 1);  // overlap w/ WMMA

        v16bf bfrag[4];
#pragma unroll
        for (int nf = 0; nf < 4; ++nf)
            bfrag[nf] = load_frag(&Bs[buf][(wn * 64 + nf * 16) * LDT], LDT);
#pragma unroll
        for (int mf = 0; mf < 4; ++mf) {
            v16bf afrag = load_frag(&As[buf][(wm * 64 + mf * 16) * LDT], LDT);
#pragma unroll
            for (int nf = 0; nf < 4; ++nf)
                acc[mf][nf] = wmma_bf16(afrag, bfrag[nf], acc[mf][nf]);
        }

        async_wait_all();    // next tile's DMA done (only un-hidden latency)
        __syncthreads();     // everyone finished reading buf
    }

    const int lane  = tid & 31;
    const int nlane = lane & 15;
    const int half  = lane >> 4;
#pragma unroll
    for (int mf = 0; mf < 4; ++mf) {
#pragma unroll
        for (int nf = 0; nf < 4; ++nf) {
            int   col = n0 + wn * 64 + nf * 16 + nlane;
            float bv  = bias ? bias[col] : 0.f;
#pragma unroll
            for (int j = 0; j < 8; ++j) {
                long  row = m0 + wm * 64 + mf * 16 + half * 8 + j;
                float v   = acc[mf][nf][j] + bv;
                if (OUT_BF16)
                    ((unsigned short*)Cout)[row * ldc + col] = f32_to_bf16(v);
                else
                    ((float*)Cout)[row * ldc + col] = v;
            }
        }
    }
}

// ---------------------------------------------------------------------------
// Flash MQA: block = 128 queries x 1 head x 1 batch; 8 waves x 16 q-rows.
// qkv layout: (B*S) x 2304 bf16, Q at col h*128, K at 2048, V at 2176.
// ---------------------------------------------------------------------------
__global__ __launch_bounds__(256) void mqa_flash_kernel(
    const unsigned short* __restrict__ qkv,   // (B*S) x 2304 bf16
    unsigned short* __restrict__ outb,        // (B*S) x 2048 bf16
    int S) {
    constexpr int LDQKV = 2304;
    constexpr int HD    = 128;
    constexpr int BQ    = 128;
    constexpr int BK    = 64;
    constexpr int KS_LD = 136;   // padded K-tile row stride
    constexpr int VT_LD = 80;    // padded V^T row stride
    constexpr int P_LD  = 80;    // padded P row stride
    const float scale = 0.08838834764831845f;  // 1/sqrt(128)

    __shared__ unsigned short Ks [BK * KS_LD];    // K tile: 64(kv) x 128(d)
    __shared__ unsigned short VTs[HD * VT_LD];    // V^T:   128(d) x 64(kv)
    __shared__ unsigned short Ps [8 * 16 * P_LD]; // per-wave P: 16 x 64

    const int tid  = threadIdx.x;
    const int wave = tid >> 5;
    const int lane = tid & 31;
    const int r    = lane & 15;
    const int half = lane >> 4;
    const int c0   = half << 3;

    const int  q0      = blockIdx.x * BQ;
    const int  head    = blockIdx.y;
    const int  b       = blockIdx.z;
    const long rowbase = (long)b * S;
    const int  qcol    = head * HD;

    // Q fragments for this wave's 16 rows live in registers for the whole loop
    v16bf qf[4];
    {
        const long qrow = rowbase + q0 + wave * 16;
#pragma unroll
        for (int kf = 0; kf < 4; ++kf) {
            const unsigned short* p =
                qkv + (qrow + r) * LDQKV + qcol + kf * 32 + c0;
            FragU u;
            u.q[0] = *(const uint4*)p;
            u.q[1] = *(const uint4*)(p + 16);
            qf[kf] = u.v;
        }
    }

    v8f acc_o[8];
#pragma unroll
    for (int d = 0; d < 8; ++d) acc_o[d] = v8f_zero();
    float mrow[8], lrow[8];
#pragma unroll
    for (int j = 0; j < 8; ++j) { mrow[j] = -3.0e30f; lrow[j] = 0.f; }

    const int nkv = (q0 + BQ) / BK;   // causal: only kv tiles up to q-tile end
    for (int t = 0; t < nkv; ++t) {
        const int kv0 = t * BK;
        __syncthreads();
        // K tile 64x128 row-major, async DMA straight into LDS
#pragma unroll
        for (int i = 0; i < 4; ++i) {
            int idx = tid + i * 256;
            int row = idx >> 4;
            int c8  = (idx & 15) << 3;
            async_copy_b128(&qkv[(rowbase + kv0 + row) * LDQKV + 2048 + c8],
                            &Ks[row * KS_LD + c8]);
        }
        // V tile 64x128, transposed into VTs[d][k] (needs VGPR round-trip)
#pragma unroll
        for (int i = 0; i < 4; ++i) {
            int   idx = tid + i * 256;
            int   k   = idx >> 4;
            int   d8  = (idx & 15) << 3;
            uint4 raw =
                *(const uint4*)&qkv[(rowbase + kv0 + k) * LDQKV + 2176 + d8];
            const unsigned short* e = (const unsigned short*)&raw;
#pragma unroll
            for (int e2 = 0; e2 < 8; ++e2)
                VTs[(d8 + e2) * VT_LD + k] = e[e2];
        }
        async_wait_all();
        __syncthreads();

        // S = Q K^T  (16 q-rows x 64 kv-cols per wave)
        v8f sacc[4];
#pragma unroll
        for (int nf = 0; nf < 4; ++nf) {
            sacc[nf] = v8f_zero();
#pragma unroll
            for (int kf = 0; kf < 4; ++kf) {
                v16bf bf = load_frag(&Ks[(nf * 16) * KS_LD + kf * 32], KS_LD);
                sacc[nf] = wmma_bf16(qf[kf], bf, sacc[nf]);
            }
        }
        // scale + causal mask
#pragma unroll
        for (int nf = 0; nf < 4; ++nf) {
            int kidx = kv0 + nf * 16 + r;
#pragma unroll
            for (int j = 0; j < 8; ++j) {
                int   qidx  = q0 + wave * 16 + half * 8 + j;
                float s     = sacc[nf][j] * scale;
                sacc[nf][j] = (kidx > qidx) ? -3.0e30f : s;
            }
        }
        // online softmax; row lives across the 16 lanes of a half-wave
        float pscale[8];
#pragma unroll
        for (int j = 0; j < 8; ++j) {
            float mx = sacc[0][j];
#pragma unroll
            for (int nf = 1; nf < 4; ++nf) mx = fmaxf(mx, sacc[nf][j]);
#pragma unroll
            for (int sh = 1; sh < 16; sh <<= 1)
                mx = fmaxf(mx, __shfl_xor(mx, sh, 32));
            float mnew = fmaxf(mrow[j], mx);
            pscale[j]  = __expf(mrow[j] - mnew);
            float sum  = 0.f;
#pragma unroll
            for (int nf = 0; nf < 4; ++nf) {
                float p     = __expf(sacc[nf][j] - mnew);
                sacc[nf][j] = p;
                sum += p;
            }
#pragma unroll
            for (int sh = 1; sh < 16; sh <<= 1)
                sum += __shfl_xor(sum, sh, 32);
            lrow[j] = lrow[j] * pscale[j] + sum;
            mrow[j] = mnew;
        }
#pragma unroll
        for (int d = 0; d < 8; ++d)
#pragma unroll
            for (int j = 0; j < 8; ++j) acc_o[d][j] *= pscale[j];

        // relayout P (C-layout regs) -> per-wave LDS as bf16 A-tile
        unsigned short* Pw = &Ps[wave * 16 * P_LD];
#pragma unroll
        for (int nf = 0; nf < 4; ++nf) {
            int col = nf * 16 + r;
#pragma unroll
            for (int j = 0; j < 8; ++j)
                Pw[(half * 8 + j) * P_LD + col] = f32_to_bf16(sacc[nf][j]);
        }
        // same-wave DS RAW: wait for all LDS stores before fragment reloads
        asm volatile("s_wait_dscnt 0" ::: "memory");

        // O += P V
#pragma unroll
        for (int kf = 0; kf < 2; ++kf) {
            v16bf pf = load_frag(&Pw[kf * 32], P_LD);
#pragma unroll
            for (int d = 0; d < 8; ++d) {
                v16bf vf =
                    load_frag(&VTs[(d * 16) * VT_LD + kf * 32], VT_LD);
                acc_o[d] = wmma_bf16(pf, vf, acc_o[d]);
            }
        }
    }

    // epilogue: O / l -> bf16 attention output
#pragma unroll
    for (int j = 0; j < 8; ++j) {
        float inv = 1.0f / lrow[j];
        long  row = rowbase + q0 + wave * 16 + half * 8 + j;
#pragma unroll
        for (int d = 0; d < 8; ++d) {
            int col = qcol + d * 16 + r;
            outb[row * 2048 + col] = f32_to_bf16(acc_o[d][j] * inv);
        }
    }
}

// ---------------------------------------------------------------------------
extern "C" void kernel_launch(void* const* d_in, const int* in_sizes, int n_in,
                              void* d_out, int out_size, void* d_ws,
                              size_t ws_size, hipStream_t stream) {
    (void)in_sizes; (void)n_in; (void)out_size; (void)ws_size;
    const float* hs  = (const float*)d_in[0];
    const float* q_w = (const float*)d_in[1];
    const float* q_b = (const float*)d_in[2];
    const float* k_w = (const float*)d_in[3];
    const float* k_b = (const float*)d_in[4];
    const float* v_w = (const float*)d_in[5];
    const float* v_b = (const float*)d_in[6];
    const float* o_w = (const float*)d_in[7];
    const float* o_b = (const float*)d_in[8];

    const int  B = 4, S = 2048, D = 2048, H = 16, HD = 128;
    const long M    = (long)B * S;   // 8192
    const int  NQKV = D + 2 * HD;    // 2304

    // workspace carve-out (256B aligned); total ~123 MB
    char*  ws  = (char*)d_ws;
    size_t off = 0;
    auto alloc = [&](size_t bytes) -> void* {
        void* p = ws + off;
        off += (bytes + 255) & ~(size_t)255;
        return p;
    };
    unsigned short* hs_bf   = (unsigned short*)alloc((size_t)M * D * 2);
    unsigned short* qkvw_bf = (unsigned short*)alloc((size_t)NQKV * D * 2);
    unsigned short* ow_bf   = (unsigned short*)alloc((size_t)D * D * 2);
    unsigned short* qkv     = (unsigned short*)alloc((size_t)M * NQKV * 2);
    unsigned short* attn    = (unsigned short*)alloc((size_t)M * D * 2);
    float*          qkv_b   = (float*)alloc((size_t)NQKV * 4);

    // 1) fp32 -> bf16 conversions (weights fused row-wise: [q_w; k_w; v_w])
    const int cgrid = 4096, cblk = 256;
    convert_f32_bf16_kernel<<<cgrid, cblk, 0, stream>>>(hs, hs_bf, M * (long)D);
    convert_f32_bf16_kernel<<<cgrid, cblk, 0, stream>>>(
        q_w, qkvw_bf, (long)D * D);
    convert_f32_bf16_kernel<<<cgrid, cblk, 0, stream>>>(
        k_w, qkvw_bf + (size_t)D * D, (long)HD * D);
    convert_f32_bf16_kernel<<<cgrid, cblk, 0, stream>>>(
        v_w, qkvw_bf + (size_t)(D + HD) * D, (long)HD * D);
    convert_f32_bf16_kernel<<<cgrid, cblk, 0, stream>>>(
        o_w, ow_bf, (long)D * D);

    // fused QKV bias (all zeros in reference, but stay general)
    (void)hipMemcpyAsync(qkv_b, q_b, (size_t)D * 4, hipMemcpyDeviceToDevice,
                         stream);
    (void)hipMemcpyAsync(qkv_b + D, k_b, (size_t)HD * 4,
                         hipMemcpyDeviceToDevice, stream);
    (void)hipMemcpyAsync(qkv_b + D + HD, v_b, (size_t)HD * 4,
                         hipMemcpyDeviceToDevice, stream);

    // 2) fused QKV projection: (8192 x 2304 x 2048) -> bf16
    gemm_bf16_kernel<true><<<dim3(NQKV / 256, M / 128), 256, 0, stream>>>(
        hs_bf, qkvw_bf, qkv_b, qkv, (int)M, NQKV, D, NQKV);

    // 3) causal MQA flash attention -> bf16
    mqa_flash_kernel<<<dim3(S / 128, H, B), 256, 0, stream>>>(qkv, attn, S);

    // 4) output projection: (8192 x 2048 x 2048) -> fp32 d_out
    gemm_bf16_kernel<false><<<dim3(D / 256, M / 128), 256, 0, stream>>>(
        attn, ow_bf, o_b, d_out, (int)M, D, D, D);
}
